// PositionExtrapolator_22265110463268
// MI455X (gfx1250) — matrix operand order repaired and checked by reading, not compiled, not run on hardware
//
#include <hip/hip_runtime.h>

// ---------------------------------------------------------------------------
// Types for CDNA5 WMMA (gfx1250, wave32)
// ---------------------------------------------------------------------------
typedef __attribute__((ext_vector_type(16))) __bf16 v16bf;
typedef __attribute__((ext_vector_type(8)))  float  v8f;

#define DIM        1024
#define HID        512
#define NB         16
#define SEQ        2048
#define TGT        4096
#define NSTEPS     2048
#define BATCH      8

// ---------------------------------------------------------------------------
// Helpers
// ---------------------------------------------------------------------------
__device__ __forceinline__ unsigned short f2bf(float x) {
    union { float f; unsigned u; } cv; cv.f = x;
    unsigned r = (cv.u + 0x7FFFu + ((cv.u >> 16) & 1u)) >> 16;
    return (unsigned short)r;
}

__device__ __forceinline__ float tanh_fast(float x) {
    x = fminf(fmaxf(x, -15.f), 15.f);
    float e = __expf(-2.0f * x);
    return (1.0f - e) / (1.0f + e);
}

__device__ __forceinline__ float gelu_fast(float x) {
    float u = 0.7978845608f * (x + 0.044715f * x * x * x);
    return 0.5f * x * (1.0f + tanh_fast(u));
}

// A fragment: 16x32 bf16 from row-major activations (ld bf16 elems).
// Rows 8..15 of the padded M=16 tile are don't-care (dup rows 0..7).
// ISA layout => VGPRs 0-3 and 4-7 each cover 8 contiguous bf16 along K:
// exactly two 16B loads per lane.
__device__ __forceinline__ v16bf load_a_bf16(const __bf16* src, int ld, int k0, int lane) {
    int row  = lane & 7;
    int half = (lane >> 4) & 1;
    union { uint4 q[2]; v16bf v; } r;
    const uint4* p = (const uint4*)src;
    int base = row * ld + k0 + (half << 4);      // bf16 index, 16B aligned
    r.q[0] = p[base >> 3];
    r.q[1] = p[(base + 8) >> 3];
    return r.v;
}

// B fragment from PRE-REPACKED weights: fragment f = 32 lanes x 32B,
// lane's 16 bf16 contiguous -> two b128 loads.
__device__ __forceinline__ v16bf load_b_frag(const uint4* p, int fragID, int lane) {
    union { uint4 q[2]; v16bf v; } r;
    const uint4* q = p + (((long long)fragID * 32 + lane) << 1);
    __builtin_prefetch((const void*)(q + 64), 0, 1);   // 1 fragment ahead
    r.q[0] = q[0];
    r.q[1] = q[1];
    return r.v;
}

__device__ __forceinline__ v8f wmma_bf16(v16bf a, v16bf b, v8f c) {
    return __builtin_amdgcn_wmma_f32_16x16x32_bf16(
        false, a, false, b, (short)0, c, false, false);
}

// ---------------------------------------------------------------------------
// Prep kernels
// ---------------------------------------------------------------------------
// Convert f32 weight (K x N row-major) into WMMA B-fragment bf16 layout:
// dword i: fragID = i>>8, lane = (i>>3)&31, v = i&7
//   n = (fragID/KB)*16 + (lane&15); k = (fragID%KB)*32 + (lane>>4)*16 + 2v
__global__ void repack_b(const float* __restrict__ W, unsigned* __restrict__ out,
                         int K, int N) {
    int i = blockIdx.x * 256 + threadIdx.x;
    int total = (K * N) >> 1;
    if (i >= total) return;
    int KB = K >> 5;
    int fragID = i >> 8, r = i & 255, lane = r >> 3, v = r & 7;
    int nb = fragID / KB, kb = fragID - nb * KB;
    int n = nb * 16 + (lane & 15);
    int k = kb * 32 + ((lane >> 4) << 4) + 2 * v;
    unsigned lo = f2bf(W[(long long)k * N + n]);
    unsigned hi = f2bf(W[(long long)(k + 1) * N + n]);
    out[i] = lo | (hi << 16);
}

// copy x (8,2048,1024) into out (8,4096,1024) front half, float4 vectorized
__global__ void copy_x_kernel(const float4* __restrict__ x, float4* __restrict__ out) {
    long long i = (long long)blockIdx.x * blockDim.x + threadIdx.x;
    const long long PER_B = (long long)SEQ * (DIM / 4);
    long long b = i / PER_B, rem = i - b * PER_B;
    out[b * ((long long)TGT * (DIM / 4)) + rem] = x[i];
}

__global__ void init_state(const float* __restrict__ x,
                           float* __restrict__ cf0,
                           unsigned short* __restrict__ cbf0,
                           unsigned short* __restrict__ cbf1) {
    int tid = blockIdx.x * blockDim.x + threadIdx.x;       // 8192
    if (tid >= BATCH * DIM) return;
    int b = tid >> 10, c = tid & (DIM - 1);
    const long long base = (long long)b * SEQ * DIM;
    float cur  = x[base + (long long)(SEQ - 1) * DIM + c];
    float prev = x[base + (long long)(SEQ - 2) * DIM + c];
    cf0[tid]  = cur;
    cbf0[tid] = f2bf(cur);
    cbf1[tid] = f2bf(prev);
}

// ---------------------------------------------------------------------------
// Step kernel 1: h = tanh([cur|prev] @ D1 + bD1)     (8x2048)x(2048x1024)
// 64 WGs (one 16-col tile each) x 4 waves split-K (16 WMMAs each) + LDS reduce
// ---------------------------------------------------------------------------
__global__ void __launch_bounds__(128)
k1_d1_tanh(const __bf16* __restrict__ curbf, const __bf16* __restrict__ prevbf,
           const uint4* __restrict__ D1f, const float* __restrict__ bD1,
           unsigned short* __restrict__ hbf) {
    __shared__ float red[4 * 16 * 8];
    int lane = threadIdx.x & 31, wave = threadIdx.x >> 5;
    int nb = blockIdx.x;                          // 0..63
    const int KB = 64;                            // 2048/32
    v8f acc = {0.f, 0.f, 0.f, 0.f, 0.f, 0.f, 0.f, 0.f};
    #pragma unroll 4
    for (int i = 0; i < 16; ++i) {
        int kb = wave * 16 + i;
        int k0 = kb * 32;
        const __bf16* asrc = (k0 < DIM) ? curbf : prevbf;  // uniform per wave
        v16bf a = load_a_bf16(asrc, DIM, k0 & (DIM - 1), lane);
        v16bf b = load_b_frag(D1f, nb * KB + kb, lane);
        acc = wmma_bf16(a, b, acc);
    }
    if (lane < 16) {
        #pragma unroll
        for (int j = 0; j < 8; ++j)
            red[(wave * 16 + lane) * 8 + j] = acc[j];
    }
    __syncthreads();
    if (threadIdx.x < 16) {
        int col = nb * 16 + threadIdx.x;
        float bias = bD1[col];
        #pragma unroll
        for (int j = 0; j < 8; ++j) {
            float s = red[(0 * 16 + threadIdx.x) * 8 + j]
                    + red[(1 * 16 + threadIdx.x) * 8 + j]
                    + red[(2 * 16 + threadIdx.x) * 8 + j]
                    + red[(3 * 16 + threadIdx.x) * 8 + j];
            hbf[j * DIM + col] = f2bf(tanh_fast(s + bias));
        }
    }
}

// ---------------------------------------------------------------------------
// Step kernel 2: nxt0 = cur + (h @ D2 + bD2)          (8x1024)x(1024x1024)
// 64 WGs x 4 waves split-K (8 WMMAs each) + LDS reduce
// ---------------------------------------------------------------------------
__global__ void __launch_bounds__(128)
k2_d2_step(const __bf16* __restrict__ hbf, const uint4* __restrict__ D2f,
           const float* __restrict__ bD2, const float* __restrict__ cf_in,
           float* __restrict__ nxt0f, unsigned short* __restrict__ nxt0bf) {
    __shared__ float red[4 * 16 * 8];
    int lane = threadIdx.x & 31, wave = threadIdx.x >> 5;
    int nb = blockIdx.x;                          // 0..63
    const int KB = 32;                            // 1024/32
    v8f acc = {0.f, 0.f, 0.f, 0.f, 0.f, 0.f, 0.f, 0.f};
    #pragma unroll 4
    for (int i = 0; i < 8; ++i) {
        int kb = wave * 8 + i;
        v16bf a = load_a_bf16(hbf, DIM, kb * 32, lane);
        v16bf b = load_b_frag(D2f, nb * KB + kb, lane);
        acc = wmma_bf16(a, b, acc);
    }
    if (lane < 16) {
        #pragma unroll
        for (int j = 0; j < 8; ++j)
            red[(wave * 16 + lane) * 8 + j] = acc[j];
    }
    __syncthreads();
    if (threadIdx.x < 16) {
        int col = nb * 16 + threadIdx.x;
        float bias = bD2[col];
        #pragma unroll
        for (int j = 0; j < 8; ++j) {
            float s = red[(0 * 16 + threadIdx.x) * 8 + j]
                    + red[(1 * 16 + threadIdx.x) * 8 + j]
                    + red[(2 * 16 + threadIdx.x) * 8 + j]
                    + red[(3 * 16 + threadIdx.x) * 8 + j];
            int idx = j * DIM + col;
            float nf = cf_in[idx] + (s + bias);   // STEP_SIZE = 1.0
            nxt0f[idx]  = nf;
            nxt0bf[idx] = f2bf(nf);
        }
    }
}

// ---------------------------------------------------------------------------
// Step kernel 3 (single WG, 512 thr = 16 waves, 3 phases):
//   g = gelu(nxt0 @ W1 + b1)  -> LDS (bf16)
//   coefs = g @ W2 + b2       -> LDS (f32)      (wave 0)
//   nxt = nxt0 + 0.1*coefs@basis -> out[:,2048+t,:], new cur (f32 + bf16)
// ---------------------------------------------------------------------------
__global__ void __launch_bounds__(512)
k3_head(const __bf16* __restrict__ nxt0bf, const float* __restrict__ nxt0f,
        const uint4* __restrict__ W1f, const float* __restrict__ b1,
        const uint4* __restrict__ W2f, const float* __restrict__ b2,
        const float* __restrict__ basis, float* __restrict__ out,
        float* __restrict__ cf_new, unsigned short* __restrict__ cbf_new, int t) {
    __shared__ unsigned gsh_u[BATCH * HID / 2];          // 8x512 bf16 = 8KB
    __shared__ float coefS[BATCH * NB];                  // 8x16
    int lane = threadIdx.x & 31, wave = threadIdx.x >> 5;
    unsigned short* gsh = (unsigned short*)gsh_u;

    // Phase 1: W1 GEMM + gelu; 32 col-tiles over 16 waves (2 each), KB=32
    #pragma unroll
    for (int tile = 0; tile < 2; ++tile) {
        int nt = tile * 16 + wave;                // col tile 0..31
        v8f acc = {0.f, 0.f, 0.f, 0.f, 0.f, 0.f, 0.f, 0.f};
        for (int kb = 0; kb < 32; ++kb) {
            v16bf a = load_a_bf16(nxt0bf, DIM, kb * 32, lane);
            v16bf b = load_b_frag(W1f, nt * 32 + kb, lane);
            acc = wmma_bf16(a, b, acc);
        }
        if (lane < 16) {
            int col = nt * 16 + lane;
            float bias = b1[col];
            #pragma unroll
            for (int j = 0; j < 8; ++j)
                gsh[j * HID + col] = f2bf(gelu_fast(acc[j] + bias));
        }
    }
    __syncthreads();

    // Phase 2: coefs = g @ W2 + b2 (8x512 x 512x16) on wave 0 only, KB=16
    if (threadIdx.x < 32) {
        v8f acc = {0.f, 0.f, 0.f, 0.f, 0.f, 0.f, 0.f, 0.f};
        for (int kb = 0; kb < 16; ++kb) {
            v16bf a = load_a_bf16((const __bf16*)gsh, HID, kb * 32, lane);
            v16bf b = load_b_frag(W2f, kb, lane);
            acc = wmma_bf16(a, b, acc);
        }
        if (lane < 16) {
            float bias = b2[lane];
            #pragma unroll
            for (int j = 0; j < 8; ++j)
                coefS[j * NB + lane] = acc[j] + bias;
        }
    }
    __syncthreads();

    // Phase 3: rank-16 basis correction, emit output + rotate state
    for (int idx = threadIdx.x; idx < BATCH * DIM; idx += 512) {
        int b = idx >> 10, c = idx & (DIM - 1);
        float s = 0.f;
        #pragma unroll
        for (int j = 0; j < NB; ++j)
            s += coefS[b * NB + j] * basis[j * DIM + c];
        float val = nxt0f[idx] + 0.1f * s;
        out[((long long)b * TGT + (SEQ + t)) * DIM + c] = val;
        cf_new[idx]  = val;
        cbf_new[idx] = f2bf(val);
    }
}

// ---------------------------------------------------------------------------
// Host launcher
// ---------------------------------------------------------------------------
extern "C" void kernel_launch(void* const* d_in, const int* in_sizes, int n_in,
                              void* d_out, int out_size, void* d_ws, size_t ws_size,
                              hipStream_t stream) {
    const float* x     = (const float*)d_in[0];
    const float* basis = (const float*)d_in[1];
    const float* W1    = (const float*)d_in[2];
    const float* b1    = (const float*)d_in[3];
    const float* W2    = (const float*)d_in[4];
    const float* b2    = (const float*)d_in[5];
    const float* D1    = (const float*)d_in[6];
    const float* bD1   = (const float*)d_in[7];
    const float* D2    = (const float*)d_in[8];
    const float* bD2   = (const float*)d_in[9];
    float* out = (float*)d_out;

    // workspace carve-out
    size_t off = 0;
    auto alloc = [&](size_t bytes) -> void* {
        void* p = (char*)d_ws + off;
        off += (bytes + 255) & ~(size_t)255;
        return p;
    };
    uint4*          D1f    = (uint4*)alloc((size_t)2 * DIM * DIM * 2);
    uint4*          D2f    = (uint4*)alloc((size_t)DIM * DIM * 2);
    uint4*          W1f    = (uint4*)alloc((size_t)DIM * HID * 2);
    uint4*          W2f    = (uint4*)alloc((size_t)HID * NB * 2);
    unsigned short* hbf    = (unsigned short*)alloc((size_t)BATCH * DIM * 2);
    float*          nxt0f  = (float*)alloc((size_t)BATCH * DIM * 4);
    unsigned short* nxt0bf = (unsigned short*)alloc((size_t)BATCH * DIM * 2);
    float*          cf[2]  = {(float*)alloc((size_t)BATCH * DIM * 4),
                              (float*)alloc((size_t)BATCH * DIM * 4)};
    unsigned short* cbf[2] = {(unsigned short*)alloc((size_t)BATCH * DIM * 2),
                              (unsigned short*)alloc((size_t)BATCH * DIM * 2)};

    // repack weights f32 -> bf16 WMMA B-fragment layout (once per call)
    {
        int n;
        n = (2 * DIM * DIM) / 2; repack_b<<<(n + 255) / 256, 256, 0, stream>>>(D1, (unsigned*)D1f, 2 * DIM, DIM);
        n = (DIM * DIM) / 2;     repack_b<<<(n + 255) / 256, 256, 0, stream>>>(D2, (unsigned*)D2f, DIM, DIM);
        n = (DIM * HID) / 2;     repack_b<<<(n + 255) / 256, 256, 0, stream>>>(W1, (unsigned*)W1f, DIM, HID);
        n = (HID * NB) / 2;      repack_b<<<(n + 255) / 256, 256, 0, stream>>>(W2, (unsigned*)W2f, HID, NB);
    }

    // copy x into front half of out
    {
        long long n4 = (long long)BATCH * SEQ * (DIM / 4);
        copy_x_kernel<<<(unsigned)(n4 / 256), 256, 0, stream>>>((const float4*)x, (float4*)out);
    }

    // init cur/prev state
    init_state<<<(BATCH * DIM + 255) / 256, 256, 0, stream>>>(x, cf[0], cbf[0], cbf[1]);

    // 2048 sequential steps; pointer ping-pong rotates cur/prev
    for (int t = 0; t < NSTEPS; ++t) {
        const unsigned short* curbf  = cbf[t & 1];
        const unsigned short* prevbf = cbf[(t + 1) & 1];
        const float*          cf_in  = cf[t & 1];
        float*                cf_out = cf[(t + 1) & 1];
        unsigned short*       cb_out = cbf[(t + 1) & 1];

        k1_d1_tanh<<<64, 128, 0, stream>>>((const __bf16*)curbf, (const __bf16*)prevbf,
                                           D1f, bD1, hbf);
        k2_d2_step<<<64, 128, 0, stream>>>((const __bf16*)hbf, D2f,
                                           bD2, cf_in, nxt0f, nxt0bf);
        k3_head<<<1, 512, 0, stream>>>((const __bf16*)nxt0bf, nxt0f,
                                       W1f, b1, W2f, b2,
                                       basis, out, cf_out, cb_out, t);
    }
}
